// CausalSelfAttentionLayer_41755672051946
// MI455X (gfx1250) — matrix operand order repaired and checked by reading, not compiled, not run on hardware
//
#include <hip/hip_runtime.h>
#include <math.h>

// ---------------------------------------------------------------------------
// Problem constants (from reference)
// ---------------------------------------------------------------------------
#define NN   50000
#define DD   128
#define EE   800000
#define NH_  4
#define DH_  32
#define FF   512          // 4*D
#define LN_EPS 1e-5f

typedef __bf16 bf16_t;
typedef bf16_t v8bf  __attribute__((ext_vector_type(8)));
typedef bf16_t v16bf __attribute__((ext_vector_type(16)));
typedef float  v8f   __attribute__((ext_vector_type(8)));

// ---------------------------------------------------------------------------
// Conversion / fill helpers
// ---------------------------------------------------------------------------
__global__ void f32_to_bf16_kernel(const float* __restrict__ src,
                                   bf16_t* __restrict__ dst, int n) {
  int i = blockIdx.x * blockDim.x + threadIdx.x;
  if (i < n) dst[i] = (bf16_t)src[i];
}

// src is K x Nc row-major f32; dst is Nc x K row-major bf16 (transposed)
__global__ void transpose_to_bf16_kernel(const float* __restrict__ src,
                                         bf16_t* __restrict__ dst, int K, int Nc) {
  int i = blockIdx.x * blockDim.x + threadIdx.x;
  if (i < K * Nc) {
    int k = i / Nc, n = i % Nc;
    dst[(size_t)n * K + k] = (bf16_t)src[i];
  }
}

__global__ void fill_f32_kernel(float* __restrict__ p, float v, int n) {
  int i = blockIdx.x * blockDim.x + threadIdx.x;
  if (i < n) p[i] = v;
}

// ---------------------------------------------------------------------------
// bf16 WMMA GEMM: C[M x Nc] = A[M x K] @ B[K x Nc] (+bias) (+gelu)
//   A  : bf16 row-major (M x K)
//   Bt : bf16 row-major (Nc x K)  == B transposed
// One wave -> one 16x16 output tile via v_wmma_f32_16x16x32_bf16.
// 4 waves/block cover a 16x64 slab. M%16==0, Nc%64==0, K%32==0 required.
// Fragment layouts follow CDNA5 ISA 7.12.2 (wave32):
//   A: lane<16 holds M=lane, K=[0..7],[16..23] (khalf=lane>>4 shifts K by 8)
//   B: lane<16 holds N=lane, K=[0..7],[16..23] (khalf=lane>>4 shifts K by 8)
//   C/D: VGPR r -> M = r + 8*(lane>>4), N = lane&15
// ---------------------------------------------------------------------------
#define GEMM_WPB 4

__global__ __launch_bounds__(32 * GEMM_WPB)
void gemm_bf16_wmma_kernel(const bf16_t* __restrict__ A,
                           const bf16_t* __restrict__ Bt,
                           const float* __restrict__ bias,
                           float* __restrict__ Cf,
                           bf16_t* __restrict__ Cb,
                           int M, int K, int Nc, int act_gelu) {
  const int lane  = threadIdx.x & 31;
  const int wave  = threadIdx.x >> 5;
  const int mtile = blockIdx.x;
  const int ntile = blockIdx.y * GEMM_WPB + wave;
  const int l15   = lane & 15;
  const int khalf = lane >> 4;

  const bf16_t* arow = A  + (size_t)(mtile * 16 + l15) * K + khalf * 8;
  const bf16_t* brow = Bt + (size_t)(ntile * 16 + l15) * K + khalf * 8;

  v8f acc = {0.f, 0.f, 0.f, 0.f, 0.f, 0.f, 0.f, 0.f};

  for (int k0 = 0; k0 < K; k0 += 32) {
    v8bf a0 = *(const v8bf*)(arow + k0);
    v8bf a1 = *(const v8bf*)(arow + k0 + 16);
    v8bf b0 = *(const v8bf*)(brow + k0);
    v8bf b1 = *(const v8bf*)(brow + k0 + 16);
    v16bf af = __builtin_shufflevector(a0, a1, 0,1,2,3,4,5,6,7,8,9,10,11,12,13,14,15);
    v16bf bfv = __builtin_shufflevector(b0, b1, 0,1,2,3,4,5,6,7,8,9,10,11,12,13,14,15);
    acc = __builtin_amdgcn_wmma_f32_16x16x32_bf16(
        /*neg_a=*/false, af, /*neg_b=*/false, bfv,
        /*c_mod=*/(short)0, acc, /*reuse_a=*/false, /*reuse_b=*/false);
  }

  const int ccol  = ntile * 16 + l15;
  const int crow0 = mtile * 16 + khalf * 8;
  const float bv  = bias ? bias[ccol] : 0.f;
#pragma unroll
  for (int r = 0; r < 8; ++r) {
    float x = acc[r] + bv;
    if (act_gelu) x = 0.5f * x * (1.f + erff(x * 0.70710678118654752f)); // exact gelu
    size_t idx = (size_t)(crow0 + r) * Nc + ccol;
    if (Cf) Cf[idx] = x;
    if (Cb) Cb[idx] = (bf16_t)x;
  }
}

// ---------------------------------------------------------------------------
// Residual + LayerNorm: out = LN(xa + xb) * w + b. One wave per row of 128.
// ---------------------------------------------------------------------------
__global__ __launch_bounds__(256)
void ln_residual_kernel(const float* __restrict__ xa, const float* __restrict__ xb,
                        const float* __restrict__ w, const float* __restrict__ b,
                        float* __restrict__ outf, bf16_t* __restrict__ outb, int nrows) {
  const int lane = threadIdx.x & 31;
  const int row  = blockIdx.x * (blockDim.x >> 5) + (threadIdx.x >> 5);
  if (row >= nrows) return;
  const float* pa = xa + (size_t)row * DD;
  const float* pb = xb + (size_t)row * DD;
  float x[4];
  float s = 0.f;
#pragma unroll
  for (int j = 0; j < 4; ++j) {
    int c = lane + 32 * j;
    x[j] = pa[c] + pb[c];
    s += x[j];
  }
#pragma unroll
  for (int o = 16; o > 0; o >>= 1) s += __shfl_xor(s, o, 32);
  const float mu = s * (1.f / (float)DD);
  float v = 0.f;
#pragma unroll
  for (int j = 0; j < 4; ++j) { float d = x[j] - mu; v += d * d; }
#pragma unroll
  for (int o = 16; o > 0; o >>= 1) v += __shfl_xor(v, o, 32);
  const float rs = rsqrtf(v * (1.f / (float)DD) + LN_EPS);
#pragma unroll
  for (int j = 0; j < 4; ++j) {
    int c = lane + 32 * j;
    float y = (x[j] - mu) * rs * w[c] + b[c];
    if (outf) outf[(size_t)row * DD + c] = y;
    if (outb) outb[(size_t)row * DD + c] = (bf16_t)y;
  }
}

// ---------------------------------------------------------------------------
// Edge phase
// ---------------------------------------------------------------------------
__device__ inline void atomicMaxF(float* addr, float v) {
  unsigned* ua = (unsigned*)addr;
  unsigned old = *ua;
  while (__uint_as_float(old) < v) {
    unsigned assumed = old;
    old = atomicCAS(ua, assumed, __float_as_uint(v));
    if (old == assumed) break;
  }
}

// Pass 1: one thread per (edge, head): logits + segment-max (atomic)
__global__ __launch_bounds__(256)
void edge_logits_kernel(const float* __restrict__ Q, const float* __restrict__ Kf,
                        const float* __restrict__ ea,
                        const long long* __restrict__ ei,
                        const long long* __restrict__ mod,
                        const float* __restrict__ w_a, const float* __restrict__ b_a,
                        float* __restrict__ logits, float* __restrict__ m) {
  int t = blockIdx.x * blockDim.x + threadIdx.x;
  if (t >= EE * NH_) return;
  int e = t >> 2, h = t & 3;
  long long s = ei[e], d = ei[EE + e];
  const float* qp = Q  + (size_t)s * DD + h * DH_;
  const float* kp = Kf + (size_t)d * DD + h * DH_;
  float dot = 0.f;
#pragma unroll
  for (int j = 0; j < DH_; ++j) dot += qp[j] * kp[j];
  dot *= 0.17677669529663689f;  // 1/sqrt(32)
  float wv = ea[e];
  float z  = wv * w_a[h] + b_a[h];          // GAMMA == 1
  float bias = 1.f / (1.f + expf(-z));
  bool mk = (mod[s] == mod[d]);
  float lg = mk ? (dot + bias) : -1.0e9f;
  logits[t] = lg;
  atomicMaxF(&m[(size_t)s * NH_ + h], lg);
}

// Pass 2: ex = exp(logit - m[src]); segment-sum denom (atomic)
__global__ __launch_bounds__(256)
void edge_exp_kernel(const long long* __restrict__ ei, const float* __restrict__ m,
                     float* __restrict__ logits, float* __restrict__ denom) {
  int t = blockIdx.x * blockDim.x + threadIdx.x;
  if (t >= EE * NH_) return;
  int e = t >> 2, h = t & 3;
  long long s = ei[e];
  float ex = expf(logits[t] - m[(size_t)s * NH_ + h]);
  logits[t] = ex;
  atomicAdd(&denom[(size_t)s * NH_ + h], ex);
}

// Pass 3: alpha = ex/denom * mask; agg[src] += alpha * v[dst]
__global__ __launch_bounds__(256)
void edge_scatter_kernel(const long long* __restrict__ ei,
                         const long long* __restrict__ mod,
                         const float* __restrict__ V,
                         const float* __restrict__ logits,
                         const float* __restrict__ denom,
                         float* __restrict__ agg) {
  int t = blockIdx.x * blockDim.x + threadIdx.x;
  if (t >= EE * NH_) return;
  int e = t >> 2, h = t & 3;
  long long s = ei[e], d = ei[EE + e];
  if (mod[s] != mod[d]) return;   // alpha *= mask
  float dn = denom[(size_t)s * NH_ + h];
  dn = dn > 1e-16f ? dn : 1e-16f;
  float alpha = logits[t] / dn;
  const float* vp = V + (size_t)d * DD + h * DH_;
  float* ap = agg + (size_t)s * DD + h * DH_;
#pragma unroll
  for (int j = 0; j < DH_; ++j) atomicAdd(ap + j, alpha * vp[j]);
}

// ---------------------------------------------------------------------------
// Host orchestration
// ---------------------------------------------------------------------------
extern "C" void kernel_launch(void* const* d_in, const int* in_sizes, int n_in,
                              void* d_out, int out_size, void* d_ws, size_t ws_size,
                              hipStream_t stream) {
  // Inputs in setup_inputs() order
  const float* H      = (const float*)d_in[0];
  const float* ea     = (const float*)d_in[1];
  const float* q_w    = (const float*)d_in[2];
  const float* k_w    = (const float*)d_in[3];
  const float* v_w    = (const float*)d_in[4];
  const float* o_w    = (const float*)d_in[5];
  const float* w_a    = (const float*)d_in[6];
  const float* b_a    = (const float*)d_in[7];
  const float* ln1_w  = (const float*)d_in[8];
  const float* ln1_b  = (const float*)d_in[9];
  const float* f1_w   = (const float*)d_in[10];
  const float* f1_b   = (const float*)d_in[11];
  const float* f2_w   = (const float*)d_in[12];
  const float* f2_b   = (const float*)d_in[13];
  const float* ln2_w  = (const float*)d_in[14];
  const float* ln2_b  = (const float*)d_in[15];
  const long long* ei  = (const long long*)d_in[16];  // int64 (2, E)
  const long long* mod = (const long long*)d_in[17];  // int64 (N,)
  float* out = (float*)d_out;

  // ---- workspace layout (256B-aligned slabs) ----
  char* W = (char*)d_ws;
  size_t off = 0;
  auto take = [&](size_t bytes) -> char* {
    char* r = W + off;
    off += (bytes + 255) & ~(size_t)255;
    return r;
  };
  bf16_t* Hb    = (bf16_t*)take((size_t)NN * DD * 2);      // bf16 H; later reused for bf16 agg
  float*  QKV   = (float*) take((size_t)NN * 3 * DD * 4);  // Q|K|V f32; later reused as bf16 G
  bf16_t* qwT   = (bf16_t*)take((size_t)DD * DD * 2);
  bf16_t* kwT   = (bf16_t*)take((size_t)DD * DD * 2);
  bf16_t* vwT   = (bf16_t*)take((size_t)DD * DD * 2);
  bf16_t* owT   = (bf16_t*)take((size_t)DD * DD * 2);
  bf16_t* f1T   = (bf16_t*)take((size_t)FF * DD * 2);      // (512 x 128)
  bf16_t* f2T   = (bf16_t*)take((size_t)DD * FF * 2);      // (128 x 512)
  float*  logit = (float*) take((size_t)EE * NH_ * 4);     // logits -> ex
  float*  segm  = (float*) take((size_t)NN * NH_ * 4);
  float*  segd  = (float*) take((size_t)NN * NH_ * 4);
  float*  agg   = (float*) take((size_t)NN * DD * 4);      // later reused as FFN2 f32 out
  float*  T1    = (float*) take((size_t)NN * DD * 4);      // o-proj out, then H1 f32 in-place
  bf16_t* H1b   = (bf16_t*)take((size_t)NN * DD * 2);

  float*  Qf = QKV;
  float*  Kf = QKV + (size_t)NN * DD;
  float*  Vf = QKV + (size_t)2 * NN * DD;
  bf16_t* Gb   = (bf16_t*)QKV;   // gelu(H1@f1+b1) bf16 (51.2MB fits in 76.8MB slab)
  bf16_t* aggb = Hb;             // bf16 copy of agg (Hb free after QKV GEMMs)
  float*  F2o  = agg;            // FFN2 f32 out (agg free after its bf16 copy)

  // ---- init ----
  hipMemsetAsync(segd, 0, (size_t)NN * NH_ * 4, stream);
  hipMemsetAsync(agg,  0, (size_t)NN * DD * 4, stream);
  fill_f32_kernel<<<(NN * NH_ + 255) / 256, 256, 0, stream>>>(segm, -3.0e38f, NN * NH_);

  // ---- convert inputs to bf16 (weights transposed for WMMA B-frag loads) ----
  f32_to_bf16_kernel<<<((size_t)NN * DD + 255) / 256, 256, 0, stream>>>(H, Hb, NN * DD);
  transpose_to_bf16_kernel<<<(DD * DD + 255) / 256, 256, 0, stream>>>(q_w, qwT, DD, DD);
  transpose_to_bf16_kernel<<<(DD * DD + 255) / 256, 256, 0, stream>>>(k_w, kwT, DD, DD);
  transpose_to_bf16_kernel<<<(DD * DD + 255) / 256, 256, 0, stream>>>(v_w, vwT, DD, DD);
  transpose_to_bf16_kernel<<<(DD * DD + 255) / 256, 256, 0, stream>>>(o_w, owT, DD, DD);
  transpose_to_bf16_kernel<<<(DD * FF + 255) / 256, 256, 0, stream>>>(f1_w, f1T, DD, FF);
  transpose_to_bf16_kernel<<<(FF * DD + 255) / 256, 256, 0, stream>>>(f2_w, f2T, FF, DD);

  const int MT = NN / 16;  // 3125 row tiles
  dim3 blkG(32 * GEMM_WPB);

  // ---- Q, K, V projections (bf16 WMMA, f32 out) ----
  gemm_bf16_wmma_kernel<<<dim3(MT, DD / 64), blkG, 0, stream>>>(
      Hb, qwT, nullptr, Qf, nullptr, NN, DD, DD, 0);
  gemm_bf16_wmma_kernel<<<dim3(MT, DD / 64), blkG, 0, stream>>>(
      Hb, kwT, nullptr, Kf, nullptr, NN, DD, DD, 0);
  gemm_bf16_wmma_kernel<<<dim3(MT, DD / 64), blkG, 0, stream>>>(
      Hb, vwT, nullptr, Vf, nullptr, NN, DD, DD, 0);

  // ---- edge softmax-attention (3 passes) ----
  const int ET = EE * NH_;
  edge_logits_kernel<<<(ET + 255) / 256, 256, 0, stream>>>(
      Qf, Kf, ea, ei, mod, w_a, b_a, logit, segm);
  edge_exp_kernel<<<(ET + 255) / 256, 256, 0, stream>>>(ei, segm, logit, segd);
  edge_scatter_kernel<<<(ET + 255) / 256, 256, 0, stream>>>(
      ei, mod, Vf, logit, segd, agg);

  // ---- output projection: T1 = agg @ o_w ----
  f32_to_bf16_kernel<<<((size_t)NN * DD + 255) / 256, 256, 0, stream>>>(agg, aggb, NN * DD);
  gemm_bf16_wmma_kernel<<<dim3(MT, DD / 64), blkG, 0, stream>>>(
      aggb, owT, nullptr, T1, nullptr, NN, DD, DD, 0);

  // ---- H1 = LN(H + T1) (f32 in-place into T1, plus bf16 copy) ----
  ln_residual_kernel<<<(NN + 7) / 8, 256, 0, stream>>>(H, T1, ln1_w, ln1_b, T1, H1b, NN);

  // ---- FFN: G = gelu(H1 @ f1 + b1) [bf16]; F2o = G @ f2 + b2 [f32] ----
  gemm_bf16_wmma_kernel<<<dim3(MT, FF / 64), blkG, 0, stream>>>(
      H1b, f1T, f1_b, nullptr, Gb, NN, DD, FF, 1);
  gemm_bf16_wmma_kernel<<<dim3(MT, DD / 64), blkG, 0, stream>>>(
      Gb, f2T, f2_b, F2o, nullptr, NN, FF, DD, 0);

  // ---- H2 = LN(H1 + ffn) -> d_out ----
  ln_residual_kernel<<<(NN + 7) / 8, 256, 0, stream>>>(T1, F2o, ln2_w, ln2_b, out, nullptr, NN);

  (void)in_sizes; (void)n_in; (void)out_size; (void)ws_size;
}
